// ScaledDotProductAttention_41472204210590
// MI455X (gfx1250) — matrix operand order repaired and checked by reading, not compiled, not run on hardware
//
#include <hip/hip_runtime.h>

// Causal scaled-dot-product attention for MI455X (gfx1250), wave32 + WMMA.
// B=8, S=2048, D=128, fp32 in/out. Compute-bound -> f16 WMMA 16x16x32 path,
// f32 accumulation. S^T trick keeps softmax + P-packing entirely in-lane.
// 64-key chunks amortize softmax overhead; causal masking only on the single
// diagonal chunk per tile. Causal load balancing: q-tile pair (p, 127-p)
// split across 2 waves with an LDS log-sum-exp merge.

#define B_ 8
#define S_ 2048
#define D_ 128

typedef _Float16 half_t;
typedef __attribute__((ext_vector_type(16))) _Float16 v16h;
typedef __attribute__((ext_vector_type(8)))  _Float16 v8h;
typedef __attribute__((ext_vector_type(8)))  float    v8f;
typedef __attribute__((ext_vector_type(4)))  float    v4f;

// (1/sqrt(128)) * log2(e): scale folded into Qh so scores are in log2 units
static constexpr float kScaleLog2e = 0.12751791892896452f;
// 10000 * log2(e): additive mask penalty in log2 units
static constexpr float kNeg2 = 14426.950408889634f;

// ---------------------------------------------------------------------------
// Prepass: Qh = f16(Q*scale*log2e); Kh = f16(K); Vt[b][d][k] = f16(V[b][k][d]);
// kbias[b][k] = mask ? 0 : -NEG2.  One-time ~25MB read, trivially fast.
// ---------------------------------------------------------------------------
__global__ __launch_bounds__(256) void prep_kernel(
    const float* __restrict__ Q, const float* __restrict__ K,
    const float* __restrict__ V, const unsigned char* __restrict__ mask,
    half_t* __restrict__ Qh, half_t* __restrict__ Kh,
    half_t* __restrict__ Vt, float* __restrict__ kbias) {
  int i = blockIdx.x * blockDim.x + threadIdx.x;
  const int total = B_ * S_ * D_;
  if (i < total) {
    Qh[i] = (half_t)(Q[i] * kScaleLog2e);
    Kh[i] = (half_t)(K[i]);
    int b = i >> 18;              // / (S_*D_)
    int r = i & (S_ * D_ - 1);
    int k = r >> 7;               // / D_
    int d = r & (D_ - 1);
    Vt[(((size_t)b * D_ + d) << 11) + k] = (half_t)(V[i]);   // <<11 == *S_
  }
  if (i < B_ * S_) {
    kbias[i] = mask[i] ? 0.0f : -kNeg2;
  }
}

// ---------------------------------------------------------------------------
// Per-wave helpers
// ---------------------------------------------------------------------------
__device__ __forceinline__ void load_qfrag(const half_t* __restrict__ Qh,
                                           int bS_plus_q, int hi, v16h qb[4]) {
  const half_t* qrow = Qh + ((size_t)bS_plus_q << 7);
#pragma unroll
  for (int c = 0; c < 4; ++c)
    qb[c] = *(const v16h*)(qrow + c * 32 + hi * 16);
}

// One 64-key chunk: 16 WMMA (S^T = K x Q^T, 4 key-tiles x 4 contraction steps)
// + online softmax + 16 WMMA (O += P x V). CAUSAL only on the diagonal chunk.
template <bool CAUSAL>
__device__ __forceinline__ void fa_chunk64(
    const half_t* __restrict__ Kbase, const half_t* __restrict__ Vbase,
    const float* __restrict__ biasbase, const v16h qb[4],
    int q, int ln, int hi, int kb,
    v8f o[8], float& m, float& ssum) {
  // ---- scores: 4 S^T tiles (16 keys x 16 queries each)
  v8f s[4];
#pragma unroll
  for (int t = 0; t < 4; ++t) s[t] = (v8f){};
#pragma unroll
  for (int c = 0; c < 4; ++c) {
#pragma unroll
    for (int t = 0; t < 4; ++t) {
      // A 16x32 layout: halves 0..15 = dims off..off+15 (off = c*32 + hi*16)
      union { v16h v; v8h h[2]; } a;
      const half_t* kp = Kbase + ((size_t)(kb + 16 * t + ln) << 7) + c * 32 + hi * 16;
      a.h[0] = *(const v8h*)(kp);
      a.h[1] = *(const v8h*)(kp + 8);
      s[t] = __builtin_amdgcn_wmma_f32_16x16x32_f16(false, a.v, false, qb[c],
                                                    (short)0, s[t], false, false);
    }
  }

  // ---- padding-mask bias (+ causal compare only on the diagonal chunk)
#pragma unroll
  for (int t = 0; t < 4; ++t) {
    const int kbt = kb + 16 * t + (hi << 3);   // first key this lane owns
    v4f b0 = *(const v4f*)(biasbase + kbt);
    v4f b1 = *(const v4f*)(biasbase + kbt + 4);
#pragma unroll
    for (int r = 0; r < 8; ++r) {
      float bv = (r < 4) ? b0[r] : b1[r - 4];
      if (CAUSAL) bv += ((kbt + r > q) ? -kNeg2 : 0.0f);
      s[t][r] += bv;
    }
  }

  // ---- online softmax (per-lane stats = per query; combine halves xor-16)
  float cm = -1e30f;
#pragma unroll
  for (int t = 0; t < 4; ++t)
#pragma unroll
    for (int r = 0; r < 8; ++r) cm = fmaxf(cm, s[t][r]);
  cm = fmaxf(cm, __shfl_xor(cm, 16, 32));
  float mnew = fmaxf(m, cm);
  float corr = exp2f(m - mnew);

  // P lands directly in A-fragment layout (two 16x32 fragments)
  v16h pa0, pa1;
  float rowsum = 0.0f;
#pragma unroll
  for (int r = 0; r < 8; ++r) {
    float p0 = exp2f(s[0][r] - mnew);
    float p1 = exp2f(s[1][r] - mnew);
    float p2 = exp2f(s[2][r] - mnew);
    float p3 = exp2f(s[3][r] - mnew);
    pa0[r] = (half_t)p0; pa0[8 + r] = (half_t)p1;
    pa1[r] = (half_t)p2; pa1[8 + r] = (half_t)p3;
    rowsum += (p0 + p1) + (p2 + p3);
  }
  rowsum += __shfl_xor(rowsum, 16, 32);
  ssum = ssum * corr + rowsum;
  m = mnew;

  // ---- broadcast per-query rescale to O rows (query = r + 8*hi)
  float fac[8];
#pragma unroll
  for (int r = 0; r < 8; ++r) fac[r] = __shfl(corr, r + (hi << 3), 32);
#pragma unroll
  for (int n = 0; n < 8; ++n)
#pragma unroll
    for (int r = 0; r < 8; ++r) o[n][r] *= fac[r];

  // ---- O += P x V   (V^T rows contiguous -> 16B fragment loads)
#pragma unroll
  for (int n = 0; n < 8; ++n) {
    const half_t* vp = Vbase + ((size_t)((n << 4) + ln) << 11) + kb + (hi << 3);
    union { v16h v; v8h h[2]; } vb0, vb1;
    vb0.h[0] = *(const v8h*)(vp);
    vb0.h[1] = *(const v8h*)(vp + 16);
    vb1.h[0] = *(const v8h*)(vp + 32);
    vb1.h[1] = *(const v8h*)(vp + 48);
    o[n] = __builtin_amdgcn_wmma_f32_16x16x32_f16(false, pa0, false, vb0.v,
                                                  (short)0, o[n], false, false);
    o[n] = __builtin_amdgcn_wmma_f32_16x16x32_f16(false, pa1, false, vb1.v,
                                                  (short)0, o[n], false, false);
  }
}

// Chunk range [c0, c1) with exactly one possible diagonal chunk at cDiag.
__device__ __forceinline__ void fa_range(
    const half_t* __restrict__ Kbase, const half_t* __restrict__ Vbase,
    const float* __restrict__ biasbase, const v16h qb[4],
    int q, int ln, int hi, int c0, int c1, int cDiag,
    v8f o[8], float& m, float& ssum) {
  const int cf = (c1 < cDiag) ? c1 : cDiag;
  for (int ch = c0; ch < cf; ++ch)
    fa_chunk64<false>(Kbase, Vbase, biasbase, qb, q, ln, hi, ch << 6, o, m, ssum);
  if (c1 > cDiag)
    fa_chunk64<true>(Kbase, Vbase, biasbase, qb, q, ln, hi, cDiag << 6, o, m, ssum);
}

// ---------------------------------------------------------------------------
// Main kernel: 8 waves/block = 4 tile-pairs/block. Pair (p, 127-p) per batch:
// role 0: short tile fully + long-tile chunks [0,h);  role 1: long [h,nL).
// Long-tile partials merged in LDS with log-sum-exp combine.
// ---------------------------------------------------------------------------
__global__ __launch_bounds__(256) void fa_kernel(
    const half_t* __restrict__ Qh, const half_t* __restrict__ Kh,
    const half_t* __restrict__ Vt, const float* __restrict__ kbias,
    float* __restrict__ out) {
  // Per-pair LDS: 2048 floats (16x128 partial O) + 32 m + 32 ssum
  __shared__ float smem[4][2112];

  const int lane = threadIdx.x & 31;
  const int ln = lane & 15;
  const int hi = lane >> 4;

  // wave-uniform indices forced into SGPRs -> scalar branches, EXEC stays full
  const int wave = __builtin_amdgcn_readfirstlane(threadIdx.x >> 5);
  const int role = wave & 1;
  const int pr   = wave >> 1;                       // pair slot in block: 0..3
  const int pairIdx = blockIdx.x * 4 + pr;          // 0..511
  const int b = pairIdx >> 6;                       // 64 pairs per batch
  const int p = pairIdx & 63;

  const int qS0 = p << 4;                           // short tile
  const int qL0 = (127 - p) << 4;                   // long tile
  const int nS = (p >> 2) + 1;                      // 64-key chunks, short tile
  const int nL = ((127 - p) >> 2) + 1;              // 64-key chunks, long tile
  const int h  = (nL - nS) >> 1;                    // role-0 share of long tile

  const half_t* Kbase   = Kh + ((size_t)b * S_ << 7);
  const half_t* Vbase   = Vt + ((size_t)b * D_ << 11);
  const float* biasbase = kbias + b * S_;

  v16h qb[4];
  v8f o[8];
  float m, ssum;

  if (role == 0) {
    // ---- short tile: full flash pass (diagonal = last chunk), store directly
    const int q = qS0 + ln;
    load_qfrag(Qh, b * S_ + q, hi, qb);
#pragma unroll
    for (int n = 0; n < 8; ++n) o[n] = (v8f){};
    m = -1e30f; ssum = 0.0f;
    fa_range(Kbase, Vbase, biasbase, qb, q, ln, hi, 0, nS, nS - 1, o, m, ssum);
    {
      float inv = 1.0f / ssum;
      float fac[8];
#pragma unroll
      for (int r = 0; r < 8; ++r) fac[r] = __shfl(inv, r + (hi << 3), 32);
      float* orow = out + ((size_t)(b * S_ + qS0) << 7);
#pragma unroll
      for (int n = 0; n < 8; ++n)
#pragma unroll
        for (int r = 0; r < 8; ++r) {
          int qq = r + (hi << 3);
          orow[(size_t)qq * D_ + (n << 4) + ln] = o[n][r] * fac[r];
        }
    }
    // ---- long tile, chunks [0, h): provably below the diagonal
    const int q2 = qL0 + ln;
    load_qfrag(Qh, b * S_ + q2, hi, qb);
#pragma unroll
    for (int n = 0; n < 8; ++n) o[n] = (v8f){};
    m = -1e30f; ssum = 0.0f;
    fa_range(Kbase, Vbase, biasbase, qb, q2, ln, hi, 0, h, nL - 1, o, m, ssum);
  } else {
    // ---- long tile, chunks [h, nL): owns the diagonal chunk
    const int q2 = qL0 + ln;
    load_qfrag(Qh, b * S_ + q2, hi, qb);
#pragma unroll
    for (int n = 0; n < 8; ++n) o[n] = (v8f){};
    m = -1e30f; ssum = 0.0f;
    fa_range(Kbase, Vbase, biasbase, qb, q2, ln, hi, h, nL, nL - 1, o, m, ssum);
    // spill partial to LDS for the merge
#pragma unroll
    for (int n = 0; n < 8; ++n)
#pragma unroll
      for (int r = 0; r < 8; ++r)
        smem[pr][((n << 3) + r) * 32 + lane] = o[n][r];
    smem[pr][2048 + lane] = m;
    smem[pr][2080 + lane] = ssum;
  }

  __syncthreads();

  if (role == 0) {
    // ---- log-sum-exp merge of the two long-tile partials, store result
    float m1 = smem[pr][2048 + lane];
    float s1 = smem[pr][2080 + lane];
    float mm = fmaxf(m, m1);
    float f0 = exp2f(m - mm);
    float f1 = exp2f(m1 - mm);
    float inv = 1.0f / (ssum * f0 + s1 * f1);
    float w0 = f0 * inv, w1 = f1 * inv;
    float fac0[8], fac1[8];
#pragma unroll
    for (int r = 0; r < 8; ++r) {
      fac0[r] = __shfl(w0, r + (hi << 3), 32);
      fac1[r] = __shfl(w1, r + (hi << 3), 32);
    }
    float* orow = out + ((size_t)(b * S_ + qL0) << 7);
#pragma unroll
    for (int n = 0; n < 8; ++n)
#pragma unroll
      for (int r = 0; r < 8; ++r) {
        float oe = smem[pr][((n << 3) + r) * 32 + lane];
        int qq = r + (hi << 3);
        orow[(size_t)qq * D_ + (n << 4) + ln] = o[n][r] * fac0[r] + oe * fac1[r];
      }
  }
}

// ---------------------------------------------------------------------------
extern "C" void kernel_launch(void* const* d_in, const int* in_sizes, int n_in,
                              void* d_out, int out_size, void* d_ws, size_t ws_size,
                              hipStream_t stream) {
  const float* Q = (const float*)d_in[0];
  const float* K = (const float*)d_in[1];
  const float* V = (const float*)d_in[2];
  const unsigned char* mask = (const unsigned char*)d_in[3];
  float* out = (float*)d_out;

  char* ws = (char*)d_ws;
  const size_t QB = (size_t)B_ * S_ * D_ * sizeof(half_t);   // 4 MiB
  const size_t need = 3 * QB + (size_t)B_ * S_ * sizeof(float);
  if (ws_size < need) return;

  half_t* Qh   = (half_t*)(ws);
  half_t* Kh   = (half_t*)(ws + QB);
  half_t* Vt   = (half_t*)(ws + 2 * QB);
  float*  kbia = (float*)(ws + 3 * QB);

  const int total = B_ * S_ * D_;
  prep_kernel<<<(total + 255) / 256, 256, 0, stream>>>(Q, K, V, mask, Qh, Kh, Vt, kbia);

  // 512 balanced tile-pairs, 4 pairs (8 waves) per block -> 128 blocks
  fa_kernel<<<128, 256, 0, stream>>>(Qh, Kh, Vt, kbia, out);
}
